// TSRNBlock_52003464020148
// MI455X (gfx1250) — matrix operand order repaired and checked by reading, not compiled, not run on hardware
//
#include <hip/hip_runtime.h>
#include <hip/hip_bf16.h>
#include <math.h>

typedef __attribute__((ext_vector_type(16))) _Float16 v16h;
typedef __attribute__((ext_vector_type(8)))  _Float16 v8h;
typedef __attribute__((ext_vector_type(8)))  float    v8f;

#define BB    4
#define TT    1024
#define DD    512
#define HH    4
#define TDIM  5
#define DHH   128
#define DFFN  256
#define DRR   512
#define MEMM  64
#define BT    (BB * TT)
#define LSTR  72   // LDS row stride in halfs (64 + 8 pad; 144B rows, 16B aligned)

// ---------------------------------------------------------------------------
// WMMA fragment helpers (layouts per CDNA5 ISA 7.12.2, wave32)
// A 16x32 f16: lanes 0-15 -> M=lane, K 0-7 (e0..7) and 16-23 (e8..15);
//              lanes 16-31 -> M=lane-16, K 8-15 and 24-31.
// B 32x16 f16: lanes 0-15 -> N=lane, K 0-7 and 16-23; lanes 16-31 -> N=lane-16, K 8-15 and 24-31.
// C 16x16 f32: n = lane&15 ; m = vgpr_r + (lane<16 ? 0 : 8).
// ---------------------------------------------------------------------------
__device__ inline v16h pack16(v8h lo, v8h hi) {
  v16h r;
#pragma unroll
  for (int e = 0; e < 8; ++e) { r[e] = lo[e]; r[e + 8] = hi[e]; }
  return r;
}

__device__ inline v16h frag_a_lds(const _Float16* tile, int lane) {
  const _Float16* row = tile + (lane & 15) * LSTR + ((lane >= 16) ? 8 : 0);
  v8h lo = *(const v8h*)row;
  v8h hi = *(const v8h*)(row + 16);
  return pack16(lo, hi);
}

__device__ inline v16h frag_b_lds(const _Float16* tile, int lane) {
  const _Float16* row = tile + (lane & 15) * LSTR + ((lane >= 16) ? 8 : 0);
  v8h lo = *(const v8h*)row;
  v8h hi = *(const v8h*)(row + 16);
  return pack16(lo, hi);
}

// ---------------------------------------------------------------------------
// 128x64-tile WMMA GEMM, K staged 64 wide.  8 waves, each owns a 32x32 patch
// -> 8 v_wmma_f32_16x16x32_f16 per LDS stage (4 accumulators).
//   C[m,n] (+)= scale * sum_k A[m,k]*B(k,n)
//   BTRANS=1 : B(k,n) = Bm[n*ldb + k]   (X @ W^T, weight stored (N,K))
//   BTRANS=0 : B(k,n) = Bm[k*ldb + n]   (X @ W)
// Optional bias[n], residual[m,n], f32 and/or f16 outputs, accumulate.
// Batched via blockIdx.z: offB/offC = (z/bdiv)*s1 + (z%bdiv)*s2 (attn ctx).
// M multiple of 128; N multiple of 64; K multiple of 64.
// ---------------------------------------------------------------------------
template <bool BTRANS, bool ACC>
__global__ void __launch_bounds__(256) gemm_wmma(
    const _Float16* __restrict__ A, int lda, long long sA,
    const _Float16* __restrict__ Bm, int ldb,
    const float* __restrict__ bias, const float* __restrict__ resid,
    float* __restrict__ C, _Float16* __restrict__ C16, int ldc,
    int K, float scale, int bdiv,
    long long sB1, long long sB2, long long sC1, long long sC2) {
  __shared__ __align__(16) _Float16 As[128 * LSTR];  // 18.4 KB
  __shared__ __align__(16) _Float16 Bs[64 * LSTR];   //  9.2 KB
  const int tid = threadIdx.x;
  const int z = blockIdx.z;
  A += (long long)z * sA;
  Bm += (long long)(z / bdiv) * sB1 + (long long)(z % bdiv) * sB2;
  const long long oc = (long long)(z / bdiv) * sC1 + (long long)(z % bdiv) * sC2;
  const int m0 = blockIdx.y * 128;
  const int n0 = blockIdx.x * 64;
  const int wave = tid >> 5, lane = tid & 31;
  const int wm = (wave & 3) * 32;   // 4 row-groups of 32
  const int wn = (wave >> 2) * 32;  // 2 col-groups of 32
  v8f acc00 = {0.f, 0.f, 0.f, 0.f, 0.f, 0.f, 0.f, 0.f};
  v8f acc01 = acc00, acc10 = acc00, acc11 = acc00;

  for (int k0 = 0; k0 < K; k0 += 64) {
    __syncthreads();
#pragma unroll
    for (int q = 0; q < 8; ++q) {  // A: 128x64 halfs, 4-half vector loads
      int i = tid + q * 256;
      int m = i >> 4, kq = (i & 15) << 2;
      *(uint2*)&As[m * LSTR + kq] =
          *(const uint2*)&A[(size_t)(m0 + m) * lda + k0 + kq];
    }
    if (BTRANS) {
#pragma unroll
      for (int q = 0; q < 4; ++q) {  // B: 64x64 halfs
        int i = tid + q * 256;
        int n = i >> 4, kq = (i & 15) << 2;
        *(uint2*)&Bs[n * LSTR + kq] =
            *(const uint2*)&Bm[(size_t)(n0 + n) * ldb + k0 + kq];
      }
    } else {
#pragma unroll
      for (int q = 0; q < 16; ++q) {  // transpose K-major -> N-major in LDS
        int i = tid + q * 256;
        int n = i >> 6, kk = i & 63;
        Bs[n * LSTR + kk] = Bm[(size_t)(k0 + kk) * ldb + n0 + n];
      }
    }
    if (k0 + 64 < K)  // global_prefetch_b8 for next K slab
      __builtin_prefetch(&A[(size_t)(m0 + (tid >> 1)) * lda + k0 + 64], 0, 0);
    __syncthreads();
#pragma unroll
    for (int ks = 0; ks < 64; ks += 32) {
      v16h a0 = frag_a_lds(As + wm * LSTR + ks, lane);
      v16h a1 = frag_a_lds(As + (wm + 16) * LSTR + ks, lane);
      v16h b0 = frag_b_lds(Bs + wn * LSTR + ks, lane);
      v16h b1 = frag_b_lds(Bs + (wn + 16) * LSTR + ks, lane);
      acc00 = __builtin_amdgcn_wmma_f32_16x16x32_f16(false, a0, false, b0,
                                                     (short)0, acc00, false, false);
      acc01 = __builtin_amdgcn_wmma_f32_16x16x32_f16(false, a0, false, b1,
                                                     (short)0, acc01, false, false);
      acc10 = __builtin_amdgcn_wmma_f32_16x16x32_f16(false, a1, false, b0,
                                                     (short)0, acc10, false, false);
      acc11 = __builtin_amdgcn_wmma_f32_16x16x32_f16(false, a1, false, b1,
                                                     (short)0, acc11, false, false);
    }
  }

  const int nl = lane & 15;
  const int mb = (lane < 16) ? 0 : 8;
  const float* residz = resid ? resid + oc : nullptr;
  float* Cz = C ? C + oc : nullptr;
  _Float16* C16z = C16 ? C16 + oc : nullptr;
#pragma unroll
  for (int mi = 0; mi < 2; ++mi) {
#pragma unroll
    for (int ni = 0; ni < 2; ++ni) {
      v8f acc = (mi == 0) ? (ni == 0 ? acc00 : acc01) : (ni == 0 ? acc10 : acc11);
#pragma unroll
      for (int r = 0; r < 8; ++r) {
        int gm = m0 + wm + mi * 16 + mb + r;
        int gn = n0 + wn + ni * 16 + nl;
        float c = acc[r] * scale;
        if (bias) c += bias[gn];
        if (residz) c += residz[(size_t)gm * ldc + gn];
        size_t o = (size_t)gm * ldc + gn;
        if (ACC) {
          Cz[o] += c;
        } else {
          if (Cz) Cz[o] = c;
          if (C16z) C16z[o] = (_Float16)c;
        }
      }
    }
  }
}

// ---------------------------------------------------------------------------
// LayerNorm (one block per row of 512), writes f16 for WMMA consumers.
// ---------------------------------------------------------------------------
__global__ void __launch_bounds__(256) layernorm_k(
    const float* __restrict__ x, const float* __restrict__ g,
    const float* __restrict__ b, _Float16* __restrict__ y16) {
  __shared__ float red[256];
  const int tid = threadIdx.x;
  const float* xr = x + (size_t)blockIdx.x * DD;
  float s = 0.f;
  for (int j = tid; j < DD; j += 256) s += xr[j];
  red[tid] = s; __syncthreads();
  for (int st = 128; st > 0; st >>= 1) { if (tid < st) red[tid] += red[tid + st]; __syncthreads(); }
  const float mean = red[0] / DD; __syncthreads();
  float v = 0.f;
  for (int j = tid; j < DD; j += 256) { float d = xr[j] - mean; v += d * d; }
  red[tid] = v; __syncthreads();
  for (int st = 128; st > 0; st >>= 1) { if (tid < st) red[tid] += red[tid + st]; __syncthreads(); }
  const float inv = rsqrtf(red[0] / DD + 1e-5f);
  _Float16* yr = y16 + (size_t)blockIdx.x * DD;
  for (int j = tid; j < DD; j += 256)
    yr[j] = (_Float16)((xr[j] - mean) * inv * g[j] + b[j]);
}

// paths = sigmoid(h @ Wp^T + bp)   (N=20, too narrow for WMMA tiles)
__global__ void paths_proj_k(const _Float16* __restrict__ h16,
                             const float* __restrict__ Wp,
                             const float* __restrict__ bp,
                             float* __restrict__ paths) {
  int idx = blockIdx.x * 256 + threadIdx.x;
  if (idx >= BT * HH * TDIM) return;
  int row = idx / (HH * TDIM), n = idx % (HH * TDIM);
  const _Float16* hr = h16 + (size_t)row * DD;
  const float* wr = Wp + (size_t)n * DD;
  float d = bp[n];
  for (int k = 0; k < DD; ++k) d += (float)hr[k] * wr[k];
  paths[idx] = 1.f / (1.f + expf(-d));
}

// Path-agreement causal scores + softmax -> f16 attention weights row.
__global__ void __launch_bounds__(256) attn_scores_k(
    const float* __restrict__ paths, _Float16* __restrict__ w16) {
  __shared__ float pq[TDIM];
  __shared__ float sims[TT];
  __shared__ float red[256];
  const int tid = threadIdx.x;
  const int bid = blockIdx.x;  // (b*H + h)*T + q
  const int q = bid % TT;
  const int bh = bid / TT;
  const int h = bh % HH;
  const int b = bh / HH;
  if (tid < TDIM) pq[tid] = paths[(size_t)(b * TT + q) * (HH * TDIM) + h * TDIM + tid];
  __syncthreads();
  float lmax = -1e30f;
  for (int k = tid; k <= q; k += 256) {
    const float* pk = paths + (size_t)(b * TT + k) * (HH * TDIM) + h * TDIM;
    float cum = 1.f, s = 0.f;
#pragma unroll
    for (int d = 0; d < TDIM; ++d) {
      float a = pq[d] * pk[d] + (1.f - pq[d]) * (1.f - pk[d]);
      cum *= a; s += cum;
    }
    s *= (1.f / TDIM);
    sims[k] = s;
    lmax = fmaxf(lmax, s);
  }
  red[tid] = lmax; __syncthreads();
  for (int st = 128; st > 0; st >>= 1) { if (tid < st) red[tid] = fmaxf(red[tid], red[tid + st]); __syncthreads(); }
  const float mx = red[0]; __syncthreads();
  float ls = 0.f;
  for (int k = tid; k <= q; k += 256) ls += expf(sims[k] - mx);
  red[tid] = ls; __syncthreads();
  for (int st = 128; st > 0; st >>= 1) { if (tid < st) red[tid] += red[tid + st]; __syncthreads(); }
  const float inv = 1.f / red[0];
  _Float16* wr = w16 + (size_t)bid * TT;
  for (int k = tid; k < TT; k += 256)
    wr[k] = (k <= q) ? (_Float16)(expf(sims[k] - mx) * inv) : (_Float16)0.f;
}

// Memory-slot softmax over 64 columns.
__global__ void __launch_bounds__(64) softmax64_k(const float* __restrict__ s,
                                                  _Float16* __restrict__ w) {
  __shared__ float red[64];
  const int tid = threadIdx.x;
  const size_t row = blockIdx.x;
  float v = s[row * MEMM + tid];
  red[tid] = v; __syncthreads();
  for (int st = 32; st > 0; st >>= 1) { if (tid < st) red[tid] = fmaxf(red[tid], red[tid + st]); __syncthreads(); }
  float mx = red[0]; __syncthreads();
  float e = expf(v - mx);
  red[tid] = e; __syncthreads();
  for (int st = 32; st > 0; st >>= 1) { if (tid < st) red[tid] += red[tid + st]; __syncthreads(); }
  w[row * MEMM + tid] = (_Float16)(e / red[0]);
}

// Power iteration for spectral normalization (3 iters + final norm), 1 block.
__global__ void __launch_bounds__(512) power_iter_k(
    const float* __restrict__ W, const float* __restrict__ v0,
    const float* __restrict__ logrho, const float* __restrict__ leak,
    float* __restrict__ out /* [0]=scale, [1]=lk */) {
  __shared__ float v[DRR];
  __shared__ float nv[DRR];
  __shared__ float red[512];
  const int tid = threadIdx.x;
  v[tid] = v0[tid];
  __syncthreads();
  float norm = 0.f;
  for (int it = 0; it < 4; ++it) {
    const float* row = W + (size_t)tid * DRR;
    float d = 0.f;
    for (int k = 0; k < DRR; ++k) d += row[k] * v[k];
    nv[tid] = d;
    red[tid] = d * d;
    __syncthreads();
    for (int st = 256; st > 0; st >>= 1) { if (tid < st) red[tid] += red[tid + st]; __syncthreads(); }
    norm = sqrtf(red[0]);
    __syncthreads();
    if (it < 3) v[tid] = (norm > 0.f) ? nv[tid] / norm : nv[tid];
    __syncthreads();
  }
  if (tid == 0) {
    float rho_t = 1.5f / (1.f + expf(-logrho[0]));
    out[0] = rho_t / fmaxf(norm, 1e-6f);
    out[1] = 1.f / (1.f + expf(-leak[0]));
  }
}

__global__ void cast16_k(const float* __restrict__ s, _Float16* __restrict__ d, int n) {
  int i = blockIdx.x * 256 + threadIdx.x;
  if (i < n) d[i] = (_Float16)s[i];
}
__global__ void scale_cast16_k(const float* __restrict__ s, const float* __restrict__ sc,
                               _Float16* __restrict__ d, int n) {
  int i = blockIdx.x * 256 + threadIdx.x;
  if (i < n) d[i] = (_Float16)(s[i] * sc[0]);
}

// ---------------------------------------------------------------------------
// Sequential reservoir scan: 1024 dependent steps of (4x512)@(512x512)^T.
// One 8-wave workgroup; state in LDS.  f16 mirror padded to 16 rows (rows
// 4..15 stay zero) so A-fragment ds_load_b128s are branch-free -> straight
// ds_load -> global_load -> v_wmma chains, no exec-mask juggling.
// ---------------------------------------------------------------------------
__global__ void __launch_bounds__(256) reservoir_scan_k(
    const float* __restrict__ u, const _Float16* __restrict__ Ws,
    const float* __restrict__ scaleLk, _Float16* __restrict__ Hs16) {
  __shared__ __align__(16) _Float16 hs16s[16][DRR];  // 16 KB, rows 4..15 == 0
  __shared__ float hs32s[4][DRR];
  const int tid = threadIdx.x, lane = tid & 31, wave = tid >> 5;
  for (int i = tid; i < 16 * DRR; i += 256) hs16s[i >> 9][i & (DRR - 1)] = (_Float16)0.f;
  for (int i = tid; i < 4 * DRR; i += 256) hs32s[i >> 9][i & (DRR - 1)] = 0.f;
  const float lk = scaleLk[1];
  __syncthreads();
  for (int t = 0; t < TT; ++t) {
    float hnew[4][4];
#pragma unroll
    for (int i = 0; i < 4; ++i)
#pragma unroll
      for (int r = 0; r < 4; ++r) hnew[i][r] = 0.f;
#pragma unroll 1
    for (int i = 0; i < 4; ++i) {
      const int nt = (wave * 4 + i) * 16;
      v8f acc = {0.f, 0.f, 0.f, 0.f, 0.f, 0.f, 0.f, 0.f};
      for (int kk = 0; kk < DRR; kk += 32) {
        const _Float16* arow = &hs16s[lane & 15][kk + ((lane >= 16) ? 8 : 0)];
        v16h a = pack16(*(const v8h*)arow, *(const v8h*)(arow + 16));
        const _Float16* brow =
            Ws + (size_t)(nt + (lane & 15)) * DRR + kk + ((lane >= 16) ? 8 : 0);
        v16h bf = pack16(*(const v8h*)brow, *(const v8h*)(brow + 16));
        acc = __builtin_amdgcn_wmma_f32_16x16x32_f16(false, a, false, bf,
                                                     (short)0, acc, false, false);
      }
      if (lane < 16) {
        const int n = nt + lane;
#pragma unroll
        for (int r = 0; r < 4; ++r) {
          float pre = acc[r] + u[((size_t)r * TT + t) * DRR + n];
          hnew[i][r] = (1.f - lk) * hs32s[r][n] + lk * tanhf(pre);
        }
      }
    }
    __syncthreads();  // all reads of old state done
    if (lane < 16) {
#pragma unroll
      for (int i = 0; i < 4; ++i) {
        const int n = (wave * 4 + i) * 16 + lane;
#pragma unroll
        for (int r = 0; r < 4; ++r) {
          float v = hnew[i][r];
          hs32s[r][n] = v;
          hs16s[r][n] = (_Float16)v;
          Hs16[((size_t)r * TT + t) * DRR + n] = (_Float16)v;
        }
      }
    }
    __syncthreads();
  }
}

// FFN complex-product * gate fusion -> f16 input for out-projection GEMM.
__global__ void ffn_combine_k(const float* __restrict__ r32,
                              const float* __restrict__ i32,
                              const float* __restrict__ g32,
                              _Float16* __restrict__ y16) {
  int idx = blockIdx.x * 256 + threadIdx.x;
  if (idx >= BT * DFFN) return;
  int row = idx / DFFN, j = idx % DFFN;
  float r = r32[idx], ii = i32[idx];
  float g1 = 1.f / (1.f + expf(-g32[(size_t)row * DD + j]));
  float g2 = 1.f / (1.f + expf(-g32[(size_t)row * DD + DFFN + j]));
  y16[(size_t)row * DD + j] = (_Float16)((r * r - ii * ii) * g1);
  y16[(size_t)row * DD + DFFN + j] = (_Float16)(2.f * r * ii * g2);
}

// Y = Rx - shifted(x)  (f16 for the lap GEMM)
__global__ void sheaf_diff_k(const float* __restrict__ Rx,
                             const float* __restrict__ x4,
                             _Float16* __restrict__ Y16, int ad) {
  int idx = blockIdx.x * 256 + threadIdx.x;
  if (idx >= BT * DD) return;
  int row = idx / DD, n = idx % DD;
  int t = row % TT;
  float nb = (t >= ad) ? x4[(size_t)(row - ad) * DD + n] : 0.f;
  Y16[idx] = (_Float16)(Rx[idx] - nb);
}

__global__ void sheaf_final_k(const float* __restrict__ x4,
                              const float* __restrict__ lap,
                              const float* __restrict__ alpha,
                              float* __restrict__ out) {
  int idx = blockIdx.x * 256 + threadIdx.x;
  if (idx >= BT * DD) return;
  out[idx] = x4[idx] - fabsf(alpha[0]) * lap[idx];
}

// ---------------------------------------------------------------------------
extern "C" void kernel_launch(void* const* d_in, const int* in_sizes, int n_in,
                              void* d_out, int out_size, void* d_ws, size_t ws_size,
                              hipStream_t stream) {
  (void)in_sizes; (void)n_in; (void)out_size; (void)ws_size;
  const float* x         = (const float*)d_in[0];
  const float* ln1_g     = (const float*)d_in[1];
  const float* ln1_b     = (const float*)d_in[2];
  const float* ln2_g     = (const float*)d_in[3];
  const float* ln2_b     = (const float*)d_in[4];
  const float* ln3_g     = (const float*)d_in[5];
  const float* ln3_b     = (const float*)d_in[6];
  const float* ln4_g     = (const float*)d_in[7];
  const float* ln4_b     = (const float*)d_in[8];
  const float* attn_Wp   = (const float*)d_in[9];
  const float* attn_bp   = (const float*)d_in[10];
  const float* attn_Wv   = (const float*)d_in[11];
  const float* attn_Wo   = (const float*)d_in[12];
  const float* ffn_Wr    = (const float*)d_in[13];
  const float* ffn_br    = (const float*)d_in[14];
  const float* ffn_Wi    = (const float*)d_in[15];
  const float* ffn_bi    = (const float*)d_in[16];
  const float* ffn_Wg    = (const float*)d_in[17];
  const float* ffn_bg    = (const float*)d_in[18];
  const float* ffn_Wout  = (const float*)d_in[19];
  const float* ffn_bout  = (const float*)d_in[20];
  const float* res_Win   = (const float*)d_in[21];
  const float* res_Wres  = (const float*)d_in[22];
  const float* res_logrho= (const float*)d_in[23];
  const float* res_leak  = (const float*)d_in[24];
  const float* res_rdout = (const float*)d_in[25];
  const float* res_v0    = (const float*)d_in[26];
  const float* mem_keys  = (const float*)d_in[27];
  const float* mem_vals  = (const float*)d_in[28];
  const float* mem_Wq    = (const float*)d_in[29];
  const float* mem_Wout  = (const float*)d_in[30];
  const float* mem_bout  = (const float*)d_in[31];
  const float* sheaf_Rw  = (const float*)d_in[32];
  const float* sheaf_Rb  = (const float*)d_in[33];
  const float* sheaf_al  = (const float*)d_in[34];

  // -------- workspace carve-out --------
  char* p = (char*)d_ws;
  auto alloc = [&](size_t bytes) -> void* {
    void* r = (void*)p; p += (bytes + 255) & ~(size_t)255; return r;
  };
  const size_t NBT = (size_t)BT;
  _Float16* h16   = (_Float16*)alloc(NBT * DD * 2);
  _Float16* Wv16  = (_Float16*)alloc((size_t)DD * DD * 2);
  _Float16* Wo16  = (_Float16*)alloc((size_t)DD * DD * 2);
  _Float16* Win16 = (_Float16*)alloc((size_t)DD * DD * 2);
  _Float16* Rd16  = (_Float16*)alloc((size_t)DD * DD * 2);
  _Float16* Wq16  = (_Float16*)alloc((size_t)DD * DD * 2);
  _Float16* Mk16  = (_Float16*)alloc((size_t)MEMM * DD * 2);
  _Float16* Mv16  = (_Float16*)alloc((size_t)MEMM * DD * 2);
  _Float16* Mo16  = (_Float16*)alloc((size_t)DD * DD * 2);
  _Float16* Wr16  = (_Float16*)alloc((size_t)DFFN * DD * 2);
  _Float16* Wi16  = (_Float16*)alloc((size_t)DFFN * DD * 2);
  _Float16* Wg16  = (_Float16*)alloc((size_t)DD * DD * 2);
  _Float16* Wf16  = (_Float16*)alloc((size_t)DD * DD * 2);
  _Float16* Rw16  = (_Float16*)alloc((size_t)4 * DD * DD * 2);
  _Float16* Ws16  = (_Float16*)alloc((size_t)DRR * DRR * 2);
  _Float16* V16   = (_Float16*)alloc(NBT * DD * 2);
  _Float16* AW16  = (_Float16*)alloc((size_t)BB * HH * TT * TT * 2);  // 32MB, L2-resident
  _Float16* ctx16 = (_Float16*)alloc(NBT * DD * 2);
  _Float16* Hs16  = (_Float16*)alloc(NBT * DD * 2);
  _Float16* q16   = (_Float16*)alloc(NBT * DD * 2);
  _Float16* w16   = (_Float16*)alloc(NBT * MEMM * 2);
  _Float16* t16   = (_Float16*)alloc(NBT * DD * 2);
  _Float16* y16   = (_Float16*)alloc(NBT * DD * 2);
  _Float16* x416  = (_Float16*)alloc(NBT * DD * 2);
  _Float16* Yd16  = (_Float16*)alloc(NBT * DD * 2);
  float* paths32  = (float*)alloc(NBT * HH * TDIM * 4);
  float* s32      = (float*)alloc(NBT * MEMM * 4);
  float* xA       = (float*)alloc(NBT * DD * 4);
  float* xB       = (float*)alloc(NBT * DD * 4);
  float* u32      = (float*)alloc(NBT * DD * 4);   // later reused as Rx
  float* r32      = (float*)alloc(NBT * DFFN * 4);
  float* i32      = (float*)alloc(NBT * DFFN * 4);
  float* g32      = (float*)alloc(NBT * DD * 4);   // later reused as lap
  float* scaleLk  = (float*)alloc(256);
  float* Rx32 = u32;
  float* lap32 = g32;

  auto cast = [&](const float* s, _Float16* d, int n) {
    cast16_k<<<(n + 255) / 256, 256, 0, stream>>>(s, d, n);
  };
  cast(attn_Wv, Wv16, DD * DD);
  cast(attn_Wo, Wo16, DD * DD);
  cast(res_Win, Win16, DD * DD);
  cast(res_rdout, Rd16, DD * DD);
  cast(mem_Wq, Wq16, DD * DD);
  cast(mem_keys, Mk16, MEMM * DD);
  cast(mem_vals, Mv16, MEMM * DD);
  cast(mem_Wout, Mo16, DD * DD);
  cast(ffn_Wr, Wr16, DFFN * DD);
  cast(ffn_Wi, Wi16, DFFN * DD);
  cast(ffn_Wg, Wg16, DD * DD);
  cast(ffn_Wout, Wf16, DD * DD);
  cast(sheaf_Rw, Rw16, 4 * DD * DD);

  // ---- stage 1: path attention ----
  layernorm_k<<<BT, 256, 0, stream>>>(x, ln1_g, ln1_b, h16);
  paths_proj_k<<<(BT * HH * TDIM + 255) / 256, 256, 0, stream>>>(h16, attn_Wp, attn_bp, paths32);
  attn_scores_k<<<BB * HH * TT, 256, 0, stream>>>(paths32, AW16);
  gemm_wmma<true, false><<<dim3(DD / 64, BT / 128, 1), 256, 0, stream>>>(
      h16, DD, 0, Wv16, DD, nullptr, nullptr, nullptr, V16, DD, DD, 1.f, 1, 0, 0, 0, 0);
  gemm_wmma<false, false><<<dim3(DHH / 64, TT / 128, BB * HH), 256, 0, stream>>>(
      AW16, TT, (long long)TT * TT, V16, DD, nullptr, nullptr, nullptr, ctx16, DD,
      TT, 1.f, HH, (long long)TT * DD, (long long)DHH, (long long)TT * DD, (long long)DHH);
  gemm_wmma<true, false><<<dim3(DD / 64, BT / 128, 1), 256, 0, stream>>>(
      ctx16, DD, 0, Wo16, DD, nullptr, x, xA, nullptr, DD, DD, 1.f, 1, 0, 0, 0, 0);

  // ---- stage 2: spectral-normalized reservoir ----
  power_iter_k<<<1, 512, 0, stream>>>(res_Wres, res_v0, res_logrho, res_leak, scaleLk);
  scale_cast16_k<<<(DRR * DRR + 255) / 256, 256, 0, stream>>>(res_Wres, scaleLk, Ws16, DRR * DRR);
  layernorm_k<<<BT, 256, 0, stream>>>(xA, ln2_g, ln2_b, h16);
  gemm_wmma<true, false><<<dim3(DRR / 64, BT / 128, 1), 256, 0, stream>>>(
      h16, DD, 0, Win16, DD, nullptr, nullptr, u32, nullptr, DRR, DD, 1.f, 1, 0, 0, 0, 0);
  reservoir_scan_k<<<1, 256, 0, stream>>>(u32, Ws16, scaleLk, Hs16);
  gemm_wmma<true, false><<<dim3(DD / 64, BT / 128, 1), 256, 0, stream>>>(
      Hs16, DRR, 0, Rd16, DRR, nullptr, xA, xB, nullptr, DD, DRR, 1.f, 1, 0, 0, 0, 0);

  // ---- stage 3: memory attention ----
  layernorm_k<<<BT, 256, 0, stream>>>(xB, ln3_g, ln3_b, h16);
  gemm_wmma<true, false><<<dim3(DD / 64, BT / 128, 1), 256, 0, stream>>>(
      h16, DD, 0, Wq16, DD, nullptr, nullptr, nullptr, q16, DD, DD, 1.f, 1, 0, 0, 0, 0);
  gemm_wmma<true, false><<<dim3(MEMM / 64, BT / 128, 1), 256, 0, stream>>>(
      q16, DD, 0, Mk16, DD, nullptr, nullptr, s32, nullptr, MEMM,
      DD, 0.04419417382415922f /* 1/sqrt(512) */, 1, 0, 0, 0, 0);
  softmax64_k<<<BT, 64, 0, stream>>>(s32, w16);
  gemm_wmma<false, false><<<dim3(DD / 64, BT / 128, 1), 256, 0, stream>>>(
      w16, MEMM, 0, Mv16, DD, nullptr, nullptr, nullptr, t16, DD, MEMM, 1.f, 1, 0, 0, 0, 0);
  gemm_wmma<true, false><<<dim3(DD / 64, BT / 128, 1), 256, 0, stream>>>(
      t16, DD, 0, Mo16, DD, mem_bout, xB, xA, nullptr, DD, DD, 1.f, 1, 0, 0, 0, 0);

  // ---- stage 4: complex-product FFN ----
  layernorm_k<<<BT, 256, 0, stream>>>(xA, ln4_g, ln4_b, h16);
  gemm_wmma<true, false><<<dim3(DFFN / 64, BT / 128, 1), 256, 0, stream>>>(
      h16, DD, 0, Wr16, DD, ffn_br, nullptr, r32, nullptr, DFFN, DD, 1.f, 1, 0, 0, 0, 0);
  gemm_wmma<true, false><<<dim3(DFFN / 64, BT / 128, 1), 256, 0, stream>>>(
      h16, DD, 0, Wi16, DD, ffn_bi, nullptr, i32, nullptr, DFFN, DD, 1.f, 1, 0, 0, 0, 0);
  gemm_wmma<true, false><<<dim3(DD / 64, BT / 128, 1), 256, 0, stream>>>(
      h16, DD, 0, Wg16, DD, ffn_bg, nullptr, g32, nullptr, DD, DD, 1.f, 1, 0, 0, 0, 0);
  ffn_combine_k<<<(BT * DFFN + 255) / 256, 256, 0, stream>>>(r32, i32, g32, y16);
  gemm_wmma<true, false><<<dim3(DD / 64, BT / 128, 1), 256, 0, stream>>>(
      y16, DD, 0, Wf16, DD, ffn_bout, xA, xB, x416, DD, DD, 1.f, 1, 0, 0, 0, 0);

  // ---- stage 5: sheaf Laplacian ----
  for (int idx = 0; idx < 4; ++idx) {
    const int ad = 3 - idx;  // OFFSETS = (-3,-2,-1,0)
    gemm_wmma<true, false><<<dim3(DD / 64, BT / 128, 1), 256, 0, stream>>>(
        x416, DD, 0, Rw16 + (size_t)idx * DD * DD, DD, sheaf_Rb + (size_t)idx * DD,
        nullptr, Rx32, nullptr, DD, DD, 1.f, 1, 0, 0, 0, 0);
    sheaf_diff_k<<<(BT * DD + 255) / 256, 256, 0, stream>>>(Rx32, xB, Yd16, ad);
    if (idx == 0)
      gemm_wmma<false, false><<<dim3(DD / 64, BT / 128, 1), 256, 0, stream>>>(
          Yd16, DD, 0, Rw16 + (size_t)idx * DD * DD, DD, nullptr, nullptr,
          lap32, nullptr, DD, DD, 1.f, 1, 0, 0, 0, 0);
    else
      gemm_wmma<false, true><<<dim3(DD / 64, BT / 128, 1), 256, 0, stream>>>(
          Yd16, DD, 0, Rw16 + (size_t)idx * DD * DD, DD, nullptr, nullptr,
          lap32, nullptr, DD, DD, 1.f, 1, 0, 0, 0, 0);
  }
  sheaf_final_k<<<(BT * DD + 255) / 256, 256, 0, stream>>>(xB, lap32, sheaf_al, (float*)d_out);
}
